// MutationAttentionExternalVTransformed_87462714016082
// MI455X (gfx1250) — matrix operand order repaired and checked by reading, not compiled, not run on hardware
//
#include <hip/hip_runtime.h>
#include <hip/hip_bf16.h>
#include <math.h>

typedef __bf16 bf16_t;
typedef bf16_t v16bf __attribute__((ext_vector_type(16)));
typedef bf16_t v8bf  __attribute__((ext_vector_type(8)));
typedef float  v8f   __attribute__((ext_vector_type(8)));
typedef float  v4f   __attribute__((ext_vector_type(4)));
typedef unsigned int v4u __attribute__((ext_vector_type(4)));
typedef unsigned int v8u __attribute__((ext_vector_type(8)));

#define NGENES 4384
#define DM     128
#define BATCH  16
#define QTILES (NGENES / 16)   // 274
#define WAVES  4
#define SCALE  0.08838834764831845f  // 1/sqrt(128)

union U16bf { v16bf bv; v8u uv; };
union U8bf  { v8bf  bv; v4u uv; };

static __device__ __forceinline__ v16bf cat8(v8bf lo, v8bf hi) {
  return __builtin_shufflevector(lo, hi, 0,1,2,3,4,5,6,7,8,9,10,11,12,13,14,15);
}

// ---- Prep: Q = (mut @ W_in + b_in) * scale  -> bf16 [B,N,D] row-major ----
__global__ __launch_bounds__(256) void prep_q(const float* __restrict__ mut,
                                              const float* __restrict__ W_in,
                                              const float* __restrict__ b_in,
                                              bf16_t* __restrict__ Qbf) {
  int idx = blockIdx.x * 256 + threadIdx.x;
  if (idx >= BATCH * NGENES * DM) return;
  int d    = idx & (DM - 1);
  int rowi = idx >> 7;                 // b*N + n
  const float* mr = mut + (size_t)rowi * 4;
  float v = b_in[d];
  v = fmaf(mr[0], W_in[0 * DM + d], v);
  v = fmaf(mr[1], W_in[1 * DM + d], v);
  v = fmaf(mr[2], W_in[2 * DM + d], v);
  v = fmaf(mr[3], W_in[3 * DM + d], v);
  Qbf[idx] = (bf16_t)(v * SCALE);
}

// ---- Prep: Kbf = bf16(key_param) [N,D];  VT = bf16((extV@W_v + b_v)^T) [D,N] ----
__global__ __launch_bounds__(256) void prep_kv(const float* __restrict__ key_param,
                                               const float* __restrict__ extV,
                                               const float* __restrict__ W_v,
                                               const float* __restrict__ b_v,
                                               bf16_t* __restrict__ Kbf,
                                               bf16_t* __restrict__ VT) {
  int idx = blockIdx.x * 256 + threadIdx.x;
  if (idx >= NGENES * DM) return;
  int d = idx & (DM - 1);
  int n = idx >> 7;
  Kbf[idx] = (bf16_t)key_param[idx];
  float v = b_v[d];
  const float* ev = extV + (size_t)n * DM;
  #pragma unroll 8
  for (int j = 0; j < DM; ++j) v = fmaf(ev[j], W_v[j * DM + d], v);
  VT[(size_t)d * NGENES + n] = (bf16_t)v;
}

// ---- Fused flash-attention (transposed: S^T = K Q^T, O^T = V^T P^T) + mean ----
__global__ __launch_bounds__(32 * WAVES) void attn_kernel(const bf16_t* __restrict__ Qbf,
                                                          const bf16_t* __restrict__ Kbf,
                                                          const bf16_t* __restrict__ VT,
                                                          float* __restrict__ acc) {
  __shared__ float red[WAVES][16][DM];   // per-wave transpose-reduce scratch (32 KB)
  const int lane  = threadIdx.x & 31;
  const int wave  = threadIdx.x >> 5;
  const int tile  = blockIdx.x * WAVES + wave;   // exactly B*QTILES tiles
  const int b     = tile / QTILES;
  const int qt    = tile % QTILES;
  const int ln    = lane & 15;
  const int half  = lane >> 4;
  const int base0 = half ? 16 : 0;

  // Q^T B-fragments (lane = query column, contiguous 16 bf16 per chunk); resident.
  v16bf qb[4];
  const bf16_t* qrow = Qbf + (size_t)(b * NGENES + qt * 16 + ln) * DM;
  #pragma unroll
  for (int c = 0; c < 4; ++c)
    qb[c] = *(const v16bf*)(qrow + 32 * c + 16 * half);

  // O^T accumulators: o[t] element v <-> (d = t*16 + v + 8*half, q = ln)
  v8f o[8];
  #pragma unroll
  for (int t = 0; t < 8; ++t) o[t] = (v8f){0.f,0.f,0.f,0.f,0.f,0.f,0.f,0.f};
  float m = -1.0e30f;   // per-column running max (consistent across the 2 halves)
  float l = 0.0f;       // per-lane partial row-sum for column q = ln

  for (int kt = 0; kt < NGENES; kt += 32) {
    // S^T tiles: s0 -> keys kt..kt+15, s1 -> keys kt+16..kt+31 (rows = keys)
    v8f s0 = (v8f){0.f,0.f,0.f,0.f,0.f,0.f,0.f,0.f};
    v8f s1 = (v8f){0.f,0.f,0.f,0.f,0.f,0.f,0.f,0.f};
    #pragma unroll
    for (int c = 0; c < 4; ++c) {
      const bf16_t* k0 = Kbf + (size_t)(kt + ln) * DM + 32 * c + base0;
      v16bf ka0 = cat8(*(const v8bf*)k0, *(const v8bf*)(k0 + 8));
      s0 = __builtin_amdgcn_wmma_f32_16x16x32_bf16(false, ka0, false, qb[c], (short)0, s0, false, false);
      const bf16_t* k1 = Kbf + (size_t)(kt + 16 + ln) * DM + 32 * c + base0;
      v16bf ka1 = cat8(*(const v8bf*)k1, *(const v8bf*)(k1 + 8));
      s1 = __builtin_amdgcn_wmma_f32_16x16x32_bf16(false, ka1, false, qb[c], (short)0, s1, false, false);
    }

    // Per-column tile max: 15 in-lane max + ONE cross-half shuffle
    float tmax = fmaxf(s0[0], s1[0]);
    #pragma unroll
    for (int v = 1; v < 8; ++v) tmax = fmaxf(tmax, fmaxf(s0[v], s1[v]));
    tmax = fmaxf(tmax, __shfl_xor(tmax, 16, 32));

    if (__any(tmax > m)) {     // rescale only when some column's max grows
      float mn = fmaxf(m, tmax);
      float al = __expf(m - mn);   // == 1.0 for non-growing columns
      l *= al;
      #pragma unroll
      for (int t = 0; t < 8; ++t) o[t] *= al;
      m = mn;
    }

    // P^T = exp(S^T - m); pack to bf16 pairs, exchange 4 dwords across halves
    v8bf p0v, p1v;
    float rs = 0.0f;
    #pragma unroll
    for (int v = 0; v < 8; ++v) {
      float p0 = __expf(s0[v] - m);   // key kt + v + 8*half
      float p1 = __expf(s1[v] - m);   // key kt + 16 + v + 8*half
      rs += p0 + p1;
      p0v[v] = (bf16_t)p0;
      p1v[v] = (bf16_t)p1;
    }
    l += rs;

    U8bf u0, u1; u0.bv = p0v; u1.bv = p1v;
    v4u snd, rcv, lo4, hi4;
    #pragma unroll
    for (int j = 0; j < 4; ++j)
      snd[j] = half ? u0.uv[j] : u1.uv[j];
    #pragma unroll
    for (int j = 0; j < 4; ++j)
      rcv[j] = (unsigned int)__shfl_xor((int)snd[j], 16, 32);
    #pragma unroll
    for (int j = 0; j < 4; ++j) {
      lo4[j] = half ? rcv[j] : u0.uv[j];   // elements 0..7  <-> k_local 16h+0..7
      hi4[j] = half ? u1.uv[j] : rcv[j];   // elements 8..15 <-> k_local 16h+8..15
    }
    U16bf up; up.uv = __builtin_shufflevector(lo4, hi4, 0,1,2,3,4,5,6,7);
    v16bf pb = up.bv;

    // O^T += V^T_chunk (A, 16d x 32k) * P^T (B, 32k x 16q)
    #pragma unroll
    for (int t = 0; t < 8; ++t) {
      const bf16_t* vt = VT + (size_t)(t * 16 + ln) * NGENES + kt + base0;
      v16bf va = cat8(*(const v8bf*)vt, *(const v8bf*)(vt + 8));
      o[t] = __builtin_amdgcn_wmma_f32_16x16x32_bf16(false, va, false, pb, (short)0, o[t], false, false);
    }
  }

  // Finalize: full l per column, normalize, LDS transpose-reduce over the 16
  // query columns, accumulate into the [B, D] mean accumulator.
  float lf = l + __shfl_xor(l, 16, 32);
  float linv = 1.0f / lf;
  #pragma unroll
  for (int t = 0; t < 8; ++t) {
    v8f ov = o[t] * linv;               // 8 floats at d = t*16 + 8*half .. +7
    v4f lo = (v4f){ov[0], ov[1], ov[2], ov[3]};
    v4f hi = (v4f){ov[4], ov[5], ov[6], ov[7]};
    *(v4f*)(&red[wave][ln][t * 16 + 8 * half])     = lo;
    *(v4f*)(&red[wave][ln][t * 16 + 8 * half + 4]) = hi;
  }
  // Same-wave, in-order DS: safe to read back without a barrier.
  v4f sum = (v4f){0.f, 0.f, 0.f, 0.f};
  #pragma unroll
  for (int q = 0; q < 16; ++q)
    sum += *(const v4f*)(&red[wave][q][lane * 4]);
  #pragma unroll
  for (int i = 0; i < 4; ++i)
    atomicAdd(&acc[b * DM + lane * 4 + i], sum[i]);
}

// ---- Final: mean -> GELU MLP -> [16,26] ----
__global__ __launch_bounds__(256) void final_mlp(const float* __restrict__ acc,
                                                 const float* __restrict__ W1,
                                                 const float* __restrict__ b1,
                                                 const float* __restrict__ W2,
                                                 const float* __restrict__ b2,
                                                 float* __restrict__ out) {
  __shared__ float mean[BATCH * DM];
  __shared__ float h[BATCH * 64];
  int t = threadIdx.x;
  for (int i = t; i < BATCH * DM; i += 256) mean[i] = acc[i] * (1.0f / (float)NGENES);
  __syncthreads();
  for (int i = t; i < BATCH * 64; i += 256) {
    int r = i >> 6, c = i & 63;
    float v = b1[c];
    #pragma unroll 8
    for (int j = 0; j < DM; ++j) v = fmaf(mean[r * DM + j], W1[j * 64 + c], v);
    h[i] = 0.5f * v * (1.0f + erff(v * 0.7071067811865475f));  // exact GELU
  }
  __syncthreads();
  for (int i = t; i < BATCH * 26; i += 256) {
    int r = i / 26, c = i % 26;
    float v = b2[c];
    #pragma unroll 8
    for (int j = 0; j < 64; ++j) v = fmaf(h[r * 64 + j], W2[j * 26 + c], v);
    out[i] = v;
  }
}

extern "C" void kernel_launch(void* const* d_in, const int* in_sizes, int n_in,
                              void* d_out, int out_size, void* d_ws, size_t ws_size,
                              hipStream_t stream) {
  const float* mut  = (const float*)d_in[0];   // [16,4384,4]
  const float* extV = (const float*)d_in[1];   // [4384,128]
  const float* W_in = (const float*)d_in[2];   // [4,128]
  const float* b_in = (const float*)d_in[3];   // [128]
  const float* keyP = (const float*)d_in[4];   // [4384,128]
  const float* W_v  = (const float*)d_in[5];   // [128,128]
  const float* b_v  = (const float*)d_in[6];   // [128]
  const float* W1   = (const float*)d_in[7];   // [128,64]
  const float* b1   = (const float*)d_in[8];   // [64]
  const float* W2   = (const float*)d_in[9];   // [64,26]
  const float* b2   = (const float*)d_in[10];  // [26]

  // Workspace layout (all naturally aligned):
  bf16_t* Qbf = (bf16_t*)d_ws;                          // 16*4384*128 bf16
  bf16_t* Kbf = Qbf + (size_t)BATCH * NGENES * DM;      // 4384*128 bf16
  bf16_t* VT  = Kbf + (size_t)NGENES * DM;              // 128*4384 bf16
  float*  acc = (float*)(VT + (size_t)NGENES * DM);     // 16*128 f32

  hipMemsetAsync(acc, 0, (size_t)BATCH * DM * sizeof(float), stream);

  int nq = BATCH * NGENES * DM;
  prep_q<<<(nq + 255) / 256, 256, 0, stream>>>(mut, W_in, b_in, Qbf);

  int nkv = NGENES * DM;
  prep_kv<<<(nkv + 255) / 256, 256, 0, stream>>>(keyP, extV, W_v, b_v, Kbf, VT);

  attn_kernel<<<(BATCH * QTILES) / WAVES, 32 * WAVES, 0, stream>>>(Qbf, Kbf, VT, acc);

  final_mlp<<<1, 256, 0, stream>>>(acc, W1, b1, W2, b2, (float*)d_out);
}
